// FullPairformer_60730837566352
// MI455X (gfx1250) — matrix operand order repaired and checked by reading, not compiled, not run on hardware
//
#include <hip/hip_runtime.h>

// ---------------------------------------------------------------------------
// FullPairformer for MI455X (gfx1250, wave32, WMMA f32_16x16x32_f16).
//
// Key algebraic simplification: sum_j(linear(x_j)) == linear(sum_j x_j),
// so the "triangle" product reduces to a per-row scalar vector SR[n][h].
// Everything else is position-wise -> 5 streaming passes, WMMA matmuls.
//
// Input order (JAX sorted-dict pytree flatten of setup_inputs()):
//  0 pair[1,512,512,128]
//  1 final_ln.b  2 final_ln.g
//  3 trans.lin1.b  4 trans.lin1.w[128,512]  5 trans.lin2.b  6 trans.lin2.w[512,128]
//  7 trans.ln.b  8 trans.ln.g
//  9 tri_in.gate.b 10 tri_in.gate.w 11 tri_in.left.b 12 tri_in.left.w
// 13 tri_in.ln.b 14 tri_in.ln.g 15 tri_in.out.b 16 tri_in.out.w
// 17 tri_in.right.b 18 tri_in.right.w
// 19 tri_out.eq.b 20 tri_out.eq.w 21 tri_out.gate.b 22 tri_out.gate.w
// 23 tri_out.left.b 24 tri_out.left.w 25 tri_out.ln.b 26 tri_out.ln.g
// 27 tri_out.out.b 28 tri_out.out.w 29 tri_out.right.b 30 tri_out.right.w
// ---------------------------------------------------------------------------

typedef _Float16 v16h __attribute__((ext_vector_type(16)));
typedef float    v8f  __attribute__((ext_vector_type(8)));

#define NROW   512
#define NPOS   512
#define CC     128
#define TILE_P 32
#define LN_EPSF 1e-5f

// A fragment: 16x32 f16, M x K. Lane l: row = l&15, K-half = (l&16)?8:0.
// v16h elems 0..7 -> K = k0+kh..k0+kh+7 ; elems 8..15 -> K = k0+16+kh..+7.
__device__ __forceinline__ v16h load_a_frag(const _Float16* __restrict__ base,
                                            int m0, int k0, int stride) {
  const int lane = threadIdx.x & 31;
  const _Float16* p = base + (m0 + (lane & 15)) * stride + k0 + ((lane & 16) >> 1);
  v16h a;
#pragma unroll
  for (int i = 0; i < 8; ++i) { a[i] = p[i]; a[8 + i] = p[16 + i]; }
  return a;
}

// B fragment: 32x16 f16, K x N, read from W^T [dout][din] (row = out channel).
// Lane l: col = l&15, K-half = (l&16)?8:0; elems 0..7 -> K = k0+kh..+7 ; elems 8..15 -> K = k0+16+kh..+7.
__device__ __forceinline__ v16h load_b_frag(const _Float16* __restrict__ wt,
                                            int n0, int k0, int stride) {
  const int lane = threadIdx.x & 31;
  const _Float16* p = wt + (size_t)(n0 + (lane & 15)) * stride + k0 + ((lane & 16) >> 1);
  v16h b;
#pragma unroll
  for (int i = 0; i < 8; ++i) { b[i] = p[i]; b[8 + i] = p[16 + i]; }
  return b;
}

__device__ __forceinline__ v8f wmma32(v16h a, v16h b, v8f c) {
  return __builtin_amdgcn_wmma_f32_16x16x32_f16(false, a, false, b, (short)0, c,
                                                false, false);
}

// In-place LayerNorm over sY[TILE_P][129] (stride 129 kills bank conflicts).
// 8 lanes per position, shuffle reduction inside a wave32.
__device__ __forceinline__ void block_ln(float* __restrict__ sY,
                                         const float* __restrict__ g,
                                         const float* __restrict__ b) {
  const int p  = threadIdx.x >> 3;
  const int c0 = (threadIdx.x & 7) * 16;
  float s = 0.f, s2 = 0.f;
#pragma unroll
  for (int c = 0; c < 16; ++c) {
    float v = sY[p * 129 + c0 + c];
    s += v; s2 += v * v;
  }
#pragma unroll
  for (int off = 1; off < 8; off <<= 1) {
    s  += __shfl_xor(s,  off, 32);
    s2 += __shfl_xor(s2, off, 32);
  }
  const float mu  = s  * (1.f / 128.f);
  const float var = s2 * (1.f / 128.f) - mu * mu;
  const float inv = rsqrtf(var + LN_EPSF);
#pragma unroll
  for (int c = 0; c < 16; ++c) {
    const int cc = c0 + c;
    float v = sY[p * 129 + cc];
    sY[p * 129 + cc] = (v - mu) * inv * g[cc] + b[cc];
  }
}

// ---------------------------------------------------------------------------
// Weight convert: w[din][dout] f32 -> wt[dout][din] f16 (W transposed).
// ---------------------------------------------------------------------------
__global__ void cvt_wt_kernel(const float* __restrict__ w, _Float16* __restrict__ wt,
                              int din, int dout) {
  int i = blockIdx.x * blockDim.x + threadIdx.x;
  if (i >= din * dout) return;
  int o = i / din, k = i - o * din;
  wt[i] = (_Float16)w[(size_t)k * dout + o];
}

// ---------------------------------------------------------------------------
// SR[n][h] = (sum_j X[n,j,:]) @ Wr[:,h] + 512*br[h]   (fp32 throughout)
// ---------------------------------------------------------------------------
__global__ __launch_bounds__(256)
void rowsum_kernel(const float* __restrict__ X, const float* __restrict__ Wr,
                   const float* __restrict__ br, float* __restrict__ SR) {
  __shared__ float sPart[2][CC];
  __shared__ float sSum[CC];
  const int row  = blockIdx.x;
  const int c    = threadIdx.x & 127;
  const int half = threadIdx.x >> 7;
  float s = 0.f;
  for (int j = half; j < NPOS; j += 2)
    s += X[((size_t)row * NPOS + j) * CC + c];
  sPart[half][c] = s;
  __syncthreads();
  if (threadIdx.x < CC) sSum[c] = sPart[0][c] + sPart[1][c];
  __syncthreads();
  if (threadIdx.x < CC) {
    const int h = threadIdx.x;
    float acc = 512.0f * br[h];
    for (int k = 0; k < CC; ++k) acc += sSum[k] * Wr[k * CC + h];
    SR[row * CC + h] = acc;
  }
}

// ---------------------------------------------------------------------------
// Triangle-mul stage: Y = LN( ((X@Wl(+We) + b) * SR[row] * sig(X@Wg+bg)) @ Wo
//                             + bo + X )
// Block: 32 positions x 128 channels; 8 waves, wave w owns out-ch [16w,16w+16).
// ---------------------------------------------------------------------------
__global__ __launch_bounds__(256)
void tri_kernel(const float* __restrict__ X, float* __restrict__ Y,
                const float* __restrict__ SR,
                const _Float16* __restrict__ WTl, const _Float16* __restrict__ WTe,
                const _Float16* __restrict__ WTg, const _Float16* __restrict__ WTo,
                const float* __restrict__ bl, const float* __restrict__ be,
                const float* __restrict__ bg, const float* __restrict__ bo,
                const float* __restrict__ lng, const float* __restrict__ lnb,
                int hasEq) {
  __shared__ __align__(16) _Float16 sX[TILE_P * CC];
  __shared__ __align__(16) _Float16 sG[TILE_P * CC];
  __shared__ float sY[TILE_P * 129];

  const int row = blockIdx.x >> 4;              // 512/32 = 16 position tiles/row
  const int p0  = (blockIdx.x & 15) * TILE_P;
  const size_t base = ((size_t)row * NPOS + p0) * CC;

  for (int i = threadIdx.x; i < TILE_P * CC; i += 256)
    sX[i] = (_Float16)X[base + i];
  __syncthreads();

  const int wave = threadIdx.x >> 5;
  const int lane = threadIdx.x & 31;
  const int col  = lane & 15;
  const int mhi  = (lane >> 4) << 3;
  const int n0   = wave * 16;
  const int ch   = n0 + col;

  const float srow  = SR[row * CC + ch];
  const float biasL = bl[ch] + (hasEq ? be[ch] : 0.0f);
  const float biasG = bg[ch];

  v16h Bl[4], Bg[4], Be[4];
#pragma unroll
  for (int k = 0; k < 4; ++k) {
    Bl[k] = load_b_frag(WTl, n0, k * 32, CC);
    Bg[k] = load_b_frag(WTg, n0, k * 32, CC);
  }
  if (hasEq) {
#pragma unroll
    for (int k = 0; k < 4; ++k) Be[k] = load_b_frag(WTe, n0, k * 32, CC);
  }

#pragma unroll
  for (int mt = 0; mt < TILE_P / 16; ++mt) {
    v16h A[4];
#pragma unroll
    for (int k = 0; k < 4; ++k) A[k] = load_a_frag(sX, mt * 16, k * 32, CC);
    v8f accL, accG;
#pragma unroll
    for (int v = 0; v < 8; ++v) { accL[v] = biasL; accG[v] = biasG; }
#pragma unroll
    for (int k = 0; k < 4; ++k) accL = wmma32(A[k], Bl[k], accL);
    if (hasEq) {
#pragma unroll
      for (int k = 0; k < 4; ++k) accL = wmma32(A[k], Be[k], accL);
    }
#pragma unroll
    for (int k = 0; k < 4; ++k) accG = wmma32(A[k], Bg[k], accG);
#pragma unroll
    for (int v = 0; v < 8; ++v) {
      float gate = 1.0f / (1.0f + __expf(-accG[v]));
      sG[(mt * 16 + mhi + v) * CC + ch] = (_Float16)(accL[v] * srow * gate);
    }
  }
  __syncthreads();

  // out projection + residual
  v16h Bo[4];
#pragma unroll
  for (int k = 0; k < 4; ++k) Bo[k] = load_b_frag(WTo, n0, k * 32, CC);
  const float biasO = bo[ch];
#pragma unroll
  for (int mt = 0; mt < TILE_P / 16; ++mt) {
    v8f acc;
#pragma unroll
    for (int v = 0; v < 8; ++v) acc[v] = biasO;
#pragma unroll
    for (int k = 0; k < 4; ++k)
      acc = wmma32(load_a_frag(sG, mt * 16, k * 32, CC), Bo[k], acc);
#pragma unroll
    for (int v = 0; v < 8; ++v) {
      int m = mt * 16 + mhi + v;
      sY[m * 129 + ch] = acc[v] + X[base + (size_t)m * CC + ch];
    }
  }
  __syncthreads();

  block_ln(sY, lng, lnb);
  __syncthreads();

  for (int i = threadIdx.x; i < TILE_P * CC; i += 256)
    Y[base + i] = sY[(i >> 7) * 129 + (i & 127)];
}

// ---------------------------------------------------------------------------
// Transition + final LN: out = LNf( LNt( relu(X@W1+b1)@W2 + b2 + X ) )
// ---------------------------------------------------------------------------
__global__ __launch_bounds__(256)
void trans_kernel(const float* __restrict__ X, float* __restrict__ Y,
                  const _Float16* __restrict__ WT1, const _Float16* __restrict__ WT2,
                  const float* __restrict__ b1, const float* __restrict__ b2,
                  const float* __restrict__ g_tr, const float* __restrict__ b_tr,
                  const float* __restrict__ g_f, const float* __restrict__ b_f) {
  __shared__ __align__(16) _Float16 sX[TILE_P * CC];
  __shared__ __align__(16) _Float16 sT[TILE_P * 512];
  __shared__ float sY[TILE_P * 129];

  const size_t base = (size_t)blockIdx.x * (TILE_P * CC);

  for (int i = threadIdx.x; i < TILE_P * CC; i += 256)
    sX[i] = (_Float16)X[base + i];
  __syncthreads();

  const int wave = threadIdx.x >> 5;
  const int lane = threadIdx.x & 31;
  const int col  = lane & 15;
  const int mhi  = (lane >> 4) << 3;

  // lin1 (128 -> 512) + relu; wave owns out-ch [64w, 64w+64)
#pragma unroll
  for (int nt = 0; nt < 4; ++nt) {
    const int n0 = wave * 64 + nt * 16;
    v16h B[4];
#pragma unroll
    for (int k = 0; k < 4; ++k) B[k] = load_b_frag(WT1, n0, k * 32, CC);
    const float bias = b1[n0 + col];
#pragma unroll
    for (int mt = 0; mt < TILE_P / 16; ++mt) {
      v8f acc;
#pragma unroll
      for (int v = 0; v < 8; ++v) acc[v] = bias;
#pragma unroll
      for (int k = 0; k < 4; ++k)
        acc = wmma32(load_a_frag(sX, mt * 16, k * 32, CC), B[k], acc);
#pragma unroll
      for (int v = 0; v < 8; ++v) {
        float r = acc[v] > 0.f ? acc[v] : 0.f;
        sT[(mt * 16 + mhi + v) * 512 + n0 + col] = (_Float16)r;
      }
    }
  }
  __syncthreads();

  // lin2 (512 -> 128) + residual; wave owns out-ch [16w, 16w+16)
  {
    const int n0 = wave * 16;
    const float bias = b2[n0 + col];
#pragma unroll
    for (int mt = 0; mt < TILE_P / 16; ++mt) {
      v8f acc;
#pragma unroll
      for (int v = 0; v < 8; ++v) acc[v] = bias;
#pragma unroll
      for (int kc = 0; kc < 16; ++kc) {
        v16h b = load_b_frag(WT2, n0, kc * 32, 512);
        v16h a = load_a_frag(sT, mt * 16, kc * 32, 512);
        acc = wmma32(a, b, acc);
      }
#pragma unroll
      for (int v = 0; v < 8; ++v) {
        int m = mt * 16 + mhi + v;
        sY[m * 129 + n0 + col] = acc[v] + X[base + (size_t)m * CC + n0 + col];
      }
    }
  }
  __syncthreads();

  block_ln(sY, g_tr, b_tr);
  __syncthreads();
  block_ln(sY, g_f, b_f);
  __syncthreads();

  for (int i = threadIdx.x; i < TILE_P * CC; i += 256)
    Y[base + i] = sY[(i >> 7) * 129 + (i & 127)];
}

// ---------------------------------------------------------------------------
extern "C" void kernel_launch(void* const* d_in, const int* in_sizes, int n_in,
                              void* d_out, int out_size, void* d_ws, size_t ws_size,
                              hipStream_t stream) {
  (void)in_sizes; (void)n_in; (void)out_size; (void)ws_size;
  const float* pair = (const float*)d_in[0];

  // workspace layout: SR | X2 | f16 transposed weights
  float* SR = (float*)d_ws;                                  // 512*128 f32
  float* X2 = SR + (size_t)NROW * CC;                        // 512*512*128 f32
  _Float16* wtp = (_Float16*)(X2 + (size_t)NROW * NPOS * CC);
  _Float16* wt_to_l = wtp; wtp += 128 * 128;
  _Float16* wt_to_e = wtp; wtp += 128 * 128;
  _Float16* wt_to_g = wtp; wtp += 128 * 128;
  _Float16* wt_to_o = wtp; wtp += 128 * 128;
  _Float16* wt_ti_l = wtp; wtp += 128 * 128;
  _Float16* wt_ti_g = wtp; wtp += 128 * 128;
  _Float16* wt_ti_o = wtp; wtp += 128 * 128;
  _Float16* wt_t1   = wtp; wtp += 128 * 512;   // lin1: din=128, dout=512
  _Float16* wt_t2   = wtp;                     // lin2: din=512, dout=128

  struct Cv { const float* src; _Float16* dst; int din, dout; };
  const Cv cv[9] = {
      {(const float*)d_in[24], wt_to_l, 128, 128},  // tri_out.left.w
      {(const float*)d_in[20], wt_to_e, 128, 128},  // tri_out.eq.w
      {(const float*)d_in[22], wt_to_g, 128, 128},  // tri_out.gate.w
      {(const float*)d_in[28], wt_to_o, 128, 128},  // tri_out.out.w
      {(const float*)d_in[12], wt_ti_l, 128, 128},  // tri_in.left.w
      {(const float*)d_in[10], wt_ti_g, 128, 128},  // tri_in.gate.w
      {(const float*)d_in[16], wt_ti_o, 128, 128},  // tri_in.out.w
      {(const float*)d_in[4],  wt_t1,  128, 512},   // trans.lin1.w
      {(const float*)d_in[6],  wt_t2,  512, 128},   // trans.lin2.w
  };
  for (int i = 0; i < 9; ++i) {
    int n = cv[i].din * cv[i].dout;
    cvt_wt_kernel<<<(n + 255) / 256, 256, 0, stream>>>(cv[i].src, cv[i].dst,
                                                       cv[i].din, cv[i].dout);
  }

  float* X1 = (float*)d_out;  // reuse output buffer as stage-1 intermediate

  // Stage 1: tri_out (eq=True)
  rowsum_kernel<<<NROW, 256, 0, stream>>>(pair, (const float*)d_in[30],
                                          (const float*)d_in[29], SR);
  tri_kernel<<<NROW * (NPOS / TILE_P), 256, 0, stream>>>(
      pair, X1, SR, wt_to_l, wt_to_e, wt_to_g, wt_to_o,
      (const float*)d_in[23], (const float*)d_in[19],
      (const float*)d_in[21], (const float*)d_in[27],
      (const float*)d_in[26], (const float*)d_in[25], 1);

  // Stage 2: tri_in (eq=False)
  rowsum_kernel<<<NROW, 256, 0, stream>>>(X1, (const float*)d_in[18],
                                          (const float*)d_in[17], SR);
  tri_kernel<<<NROW * (NPOS / TILE_P), 256, 0, stream>>>(
      X1, X2, SR, wt_ti_l, nullptr, wt_ti_g, wt_ti_o,
      (const float*)d_in[11], nullptr,
      (const float*)d_in[9], (const float*)d_in[15],
      (const float*)d_in[14], (const float*)d_in[13], 0);

  // Stage 3: transition + final LN
  trans_kernel<<<(NROW * NPOS) / TILE_P, 256, 0, stream>>>(
      X2, (float*)d_out, wt_t1, wt_t2,
      (const float*)d_in[3], (const float*)d_in[5],
      (const float*)d_in[8], (const float*)d_in[7],
      (const float*)d_in[2], (const float*)d_in[1]);
}
